// rnn_test_3layer_40510131536144
// MI455X (gfx1250) — matrix operand order repaired and checked by reading, not compiled, not run on hardware
//
#include <hip/hip_runtime.h>

// ---------------- problem sizes ----------------
constexpr int T_STEPS = 250;
constexpr int BATCH   = 64;
constexpr int INDIM   = 2752;          // 2*32*43
constexpr int H1N     = 512;
constexpr int H2N     = 512;
constexpr int BRN     = 4;
constexpr int N2      = H2N * BRN;     // 2048
constexpr int KCAT    = H1N + H2N;     // 1024
constexpr int OUT_PAD = 112;           // 100 padded to 7*16
constexpr int OUT_N   = 100;
constexpr int NBLK    = 32;            // persistent workgroups
constexpr int NTHR    = 256;           // 8 waves (wave32) per WG
constexpr int NWAVES  = NBLK * (NTHR / 32); // 256 waves

typedef __attribute__((ext_vector_type(16))) _Float16 v16h;
typedef __attribute__((ext_vector_type(8)))  _Float16 v8h;
typedef __attribute__((ext_vector_type(8)))  float    v8f;

// ---------------- WMMA helpers ----------------
__device__ inline v8f wmma16(v16h a, v16h b, v8f c) {
  // emits v_wmma_f32_16x16x32_f16
  return __builtin_amdgcn_wmma_f32_16x16x32_f16(false, a, false, b, (short)0, c,
                                                false, false);
}

// A matrix 16x32 f16, row-major source; lane = 16 rows x (half selects K banks).
// ISA 7.12.2: lanes 0-15 hold K=[k0..k0+7] then [k0+16..k0+23]; lanes16-31 +8.
__device__ inline v16h load_a_f16(const _Float16* row, int k0, int half) {
  const _Float16* p = row + k0 + half * 8;
  v8h lo = *(const v8h*)p;          // 16B aligned
  v8h hi = *(const v8h*)(p + 16);
  v16h a;
#pragma unroll
  for (int i = 0; i < 8; ++i) { a[i] = lo[i]; a[i + 8] = hi[i]; }
  return a;
}

// same but converting from f32 on the fly (binary spike input)
__device__ inline v16h load_a_f32(const float* row, int k0, int half) {
  const float* p = row + k0 + half * 8;
  v8f lo = *(const v8f*)p;          // 32B aligned
  v8f hi = *(const v8f*)(p + 16);
  v16h a;
#pragma unroll
  for (int i = 0; i < 8; ++i) { a[i] = (_Float16)lo[i]; a[i + 8] = (_Float16)hi[i]; }
  return a;
}

// B matrix 32x16 f16: lane n holds column n; lanes0-15 K=[k0..k0+7] then
// [k0+16..k0+23]; lanes16-31 +8 (same K order as A).
__device__ inline v16h load_b_f16(const _Float16* row, int k0, int half) {
  const _Float16* p = row + k0 + half * 8;
  v8h lo = *(const v8h*)p;
  v8h hi = *(const v8h*)(p + 16);
  v16h b;
#pragma unroll
  for (int i = 0; i < 8; ++i) { b[i] = lo[i]; b[i + 8] = hi[i]; }
  return b;
}

// ---------------- grid-wide barrier (persistent kernel) ----------------
__device__ inline void grid_sync(unsigned* cnt, unsigned* gen) {
  __syncthreads();
  if (threadIdx.x == 0) {
    __threadfence();
    unsigned g = __hip_atomic_load(gen, __ATOMIC_RELAXED, __HIP_MEMORY_SCOPE_AGENT);
    unsigned old = __hip_atomic_fetch_add(cnt, 1u, __ATOMIC_ACQ_REL,
                                          __HIP_MEMORY_SCOPE_AGENT);
    if (old == (unsigned)(NBLK - 1)) {
      __hip_atomic_store(cnt, 0u, __ATOMIC_RELAXED, __HIP_MEMORY_SCOPE_AGENT);
      __hip_atomic_fetch_add(gen, 1u, __ATOMIC_RELEASE, __HIP_MEMORY_SCOPE_AGENT);
    } else {
      while (__hip_atomic_load(gen, __ATOMIC_ACQUIRE, __HIP_MEMORY_SCOPE_AGENT) == g)
        __builtin_amdgcn_s_sleep(1);
    }
    __threadfence();
  }
  __syncthreads();
}

// ---------------- prologue kernels ----------------
__global__ void k_cvt_f16(const float* __restrict__ s, _Float16* __restrict__ d, int n) {
  for (int i = blockIdx.x * blockDim.x + threadIdx.x; i < n; i += gridDim.x * blockDim.x)
    d[i] = (_Float16)s[i];
}
__global__ void k_sigmoid(const float* __restrict__ s, float* __restrict__ d, int n) {
  for (int i = blockIdx.x * blockDim.x + threadIdx.x; i < n; i += gridDim.x * blockDim.x)
    d[i] = 1.0f / (1.0f + __expf(-s[i]));
}
__global__ void k_w4pad(const float* __restrict__ w4, _Float16* __restrict__ d) {
  int i = blockIdx.x * blockDim.x + threadIdx.x;
  if (i < OUT_PAD * H2N) {
    int r = i / H2N, c = i % H2N;
    d[i] = (r < OUT_N) ? (_Float16)w4[r * H2N + c] : (_Float16)0.0f;
  }
}

// ---------------- persistent SNN kernel ----------------
__global__ __launch_bounds__(NTHR, 1) void snn_persistent(
    const float* __restrict__ dvs,                 // [B, T, INDIM] f32 (0/1)
    const float* __restrict__ b1, const float* __restrict__ b2,
    const float* __restrict__ b4,
    const _Float16* __restrict__ W1h,              // [H1, INDIM]
    const _Float16* __restrict__ W2h,              // [N2, KCAT]
    const _Float16* __restrict__ W4h,              // [OUT_PAD, H2]
    const float* __restrict__ alpha1, const float* __restrict__ alpha2,
    const float* __restrict__ beta,                // [N2]
    float* __restrict__ mem1, float* __restrict__ mem2,
    float* __restrict__ dstate,                    // [B, N2]
    float* __restrict__ accum,                     // [B, OUT_PAD]
    _Float16* __restrict__ kbuf,                   // [2, B, KCAT] spikes ping-pong
    unsigned* __restrict__ bar,
    float* __restrict__ out)
{
  const int lane = threadIdx.x & 31;
  const int half = lane >> 4;
  const int l15  = lane & 15;
  const int wid  = blockIdx.x * (NTHR / 32) + (threadIdx.x >> 5);
  unsigned* cnt = bar;
  unsigned* gen = bar + 16;

  for (int t = 0; t < T_STEPS; ++t) {
    _Float16* kb_cur = kbuf + (size_t)(t & 1) * (BATCH * KCAT);       // sp1(t) | sp2(t-1)
    _Float16* kb_nxt = kbuf + (size_t)((t + 1) & 1) * (BATCH * KCAT); // sp1(t-1) | sp2(t)

    // ---- Phase A: layer-1 LIF GEMM (128 waves) + readout of sp2(t-1) (28 waves) ----
    if (wid < 128) {
      const int m0 = (wid & 3) * 16;
      const int n0 = (wid >> 2) * 16;
      const float* arow = dvs + (size_t)(m0 + l15) * (T_STEPS * INDIM) + (size_t)t * INDIM;
      const _Float16* brow = W1h + (size_t)(n0 + l15) * INDIM;
      v8f c = {};
      for (int k0 = 0; k0 < INDIM; k0 += 32) {
        __builtin_prefetch(brow + k0 + 128, 0, 1);   // global_prefetch_b8
        v16h a = load_a_f32(arow, k0, half);
        v16h b = load_b_f16(brow, k0, half);
        c = wmma16(a, b, c);
      }
      const int n = n0 + l15;
      const float al = alpha1[n];
      const float bias = b1[n];
#pragma unroll
      for (int j = 0; j < 8; ++j) {
        const int br = m0 + j + 8 * half;            // batch row
        const float d1 = c[j] + bias;
        const size_t idx = (size_t)br * H1N + n;
        const float spp = (float)kb_nxt[(size_t)br * KCAT + n];   // sp1(t-1)
        const float nm = mem1[idx] * al + (1.0f - al) * d1 - spp;
        mem1[idx] = nm;
        kb_cur[(size_t)br * KCAT + n] =
            (_Float16)((nm - 1.0f) > 0.0f ? 1.0f : 0.0f);          // sp1(t)
      }
    } else if (wid < 156 && t > 0) {
      const int id = wid - 128;
      const int m0 = (id & 3) * 16;
      const int n0 = (id >> 2) * 16;
      const _Float16* arow = kb_cur + (size_t)(m0 + l15) * KCAT + H1N; // sp2(t-1)
      const _Float16* brow = W4h + (size_t)(n0 + l15) * H2N;
      v8f c = {};
      for (int k0 = 0; k0 < H2N; k0 += 32) {
        v16h a = load_a_f16(arow, k0, half);
        v16h b = load_b_f16(brow, k0, half);
        c = wmma16(a, b, c);
      }
      const int n = n0 + l15;
#pragma unroll
      for (int j = 0; j < 8; ++j) {
        const int br = m0 + j + 8 * half;
        accum[(size_t)br * OUT_PAD + n] += c[j];
      }
    }
    grid_sync(cnt, gen);

    // ---- Phase B: layer-2 dendritic GEMM + branch-sum + LIF (512 tiles) ----
    for (int task = wid; task < 512; task += NWAVES) {
      const int m0 = (task & 3) * 16;
      const int n0 = (task >> 2) * 16;
      const _Float16* arow = kb_cur + (size_t)(m0 + l15) * KCAT;  // [sp1(t), sp2(t-1)]
      const _Float16* brow = W2h + (size_t)(n0 + l15) * KCAT;
      v8f c = {};
      for (int k0 = 0; k0 < KCAT; k0 += 32) {
        v16h a = load_a_f16(arow, k0, half);
        v16h b = load_b_f16(brow, k0, half);
        c = wmma16(a, b, c);
      }
      const int n = n0 + l15;                 // 0..2047 = h*4 + branch
      const float bias = b2[n];
      const float bt = beta[n];
      const float al2 = alpha2[n >> 2];
#pragma unroll
      for (int j = 0; j < 8; ++j) {
        const int br = m0 + j + 8 * half;
        const size_t idx = (size_t)br * N2 + n;
        const float d2 = c[j] + bias;
        float nds = bt * dstate[idx] + (1.0f - bt) * d2;
        dstate[idx] = nds;
        // sum over 4 branches = 4 consecutive lanes (wave32 shuffle)
        float s = nds + __shfl_xor(nds, 1, 32);
        s += __shfl_xor(s, 2, 32);
        if ((lane & 3) == 0) {
          const int h = n >> 2;
          const size_t i2 = (size_t)br * H2N + h;
          const float spp = (float)kb_cur[(size_t)br * KCAT + H1N + h]; // sp2(t-1)
          const float nm = mem2[i2] * al2 + (1.0f - al2) * s - spp;
          mem2[i2] = nm;
          kb_nxt[(size_t)br * KCAT + H1N + h] =
              (_Float16)((nm - 1.0f) > 0.0f ? 1.0f : 0.0f);            // sp2(t)
        }
      }
    }
    grid_sync(cnt, gen);
  }

  // ---- final readout for sp2(T-1), which lives in kbuf[T&1] ----
  {
    const _Float16* kb = kbuf + (size_t)(T_STEPS & 1) * (BATCH * KCAT);
    if (wid < 28) {
      const int m0 = (wid & 3) * 16;
      const int n0 = (wid >> 2) * 16;
      const _Float16* arow = kb + (size_t)(m0 + l15) * KCAT + H1N;
      const _Float16* brow = W4h + (size_t)(n0 + l15) * H2N;
      v8f c = {};
      for (int k0 = 0; k0 < H2N; k0 += 32) {
        v16h a = load_a_f16(arow, k0, half);
        v16h b = load_b_f16(brow, k0, half);
        c = wmma16(a, b, c);
      }
      const int n = n0 + l15;
#pragma unroll
      for (int j = 0; j < 8; ++j) {
        const int br = m0 + j + 8 * half;
        accum[(size_t)br * OUT_PAD + n] += c[j];
      }
    }
  }
  grid_sync(cnt, gen);

  // ---- epilogue: out = accum/T + b4 ----
  {
    const int tid = blockIdx.x * NTHR + threadIdx.x;
    if (tid < BATCH * OUT_N) {
      const int b = tid / OUT_N, o = tid % OUT_N;
      out[tid] = accum[(size_t)b * OUT_PAD + o] * (1.0f / (float)T_STEPS) + b4[o];
    }
  }
}

// ---------------- launch ----------------
extern "C" void kernel_launch(void* const* d_in, const int* in_sizes, int n_in,
                              void* d_out, int out_size, void* d_ws, size_t ws_size,
                              hipStream_t stream) {
  const float* dvs = (const float*)d_in[0];
  const float* W1  = (const float*)d_in[1];
  const float* b1  = (const float*)d_in[2];
  const float* tm1 = (const float*)d_in[3];
  const float* W2  = (const float*)d_in[4];
  const float* b2  = (const float*)d_in[5];
  const float* tm2 = (const float*)d_in[6];
  const float* tn  = (const float*)d_in[7];
  const float* W4  = (const float*)d_in[8];
  const float* b4  = (const float*)d_in[9];
  float* out = (float*)d_out;

  char* base = (char*)d_ws;
  size_t off = 0;
  auto take = [&](size_t bytes) -> char* {
    char* p = base + off;
    off = (off + bytes + 255) & ~(size_t)255;
    return p;
  };
  _Float16* W1h = (_Float16*)take((size_t)H1N * INDIM * 2);
  _Float16* W2h = (_Float16*)take((size_t)N2 * KCAT * 2);
  _Float16* W4h = (_Float16*)take((size_t)OUT_PAD * H2N * 2);
  float* alpha1 = (float*)take((size_t)H1N * 4);
  float* alpha2 = (float*)take((size_t)H2N * 4);
  float* beta   = (float*)take((size_t)N2 * 4);
  char* zstart  = base + off;                     // everything below is zero-init state
  float* mem1   = (float*)take((size_t)BATCH * H1N * 4);
  float* mem2   = (float*)take((size_t)BATCH * H2N * 4);
  float* dst    = (float*)take((size_t)BATCH * N2 * 4);
  float* acc    = (float*)take((size_t)BATCH * OUT_PAD * 4);
  _Float16* kbuf = (_Float16*)take((size_t)2 * BATCH * KCAT * 2);
  unsigned* bar = (unsigned*)take(256);
  if (off > ws_size) return;                       // workspace too small: bail

  hipMemsetAsync(zstart, 0, (size_t)((base + off) - zstart), stream);
  k_cvt_f16<<<1024, 256, 0, stream>>>(W1, W1h, H1N * INDIM);
  k_cvt_f16<<<1024, 256, 0, stream>>>(W2, W2h, N2 * KCAT);
  k_w4pad<<<(OUT_PAD * H2N + 255) / 256, 256, 0, stream>>>(W4, W4h);
  k_sigmoid<<<(H1N + 255) / 256, 256, 0, stream>>>(tm1, alpha1, H1N);
  k_sigmoid<<<(H2N + 255) / 256, 256, 0, stream>>>(tm2, alpha2, H2N);
  k_sigmoid<<<(N2 + 255) / 256, 256, 0, stream>>>(tn, beta, N2);
  snn_persistent<<<NBLK, NTHR, 0, stream>>>(dvs, b1, b2, b4, W1h, W2h, W4h,
                                            alpha1, alpha2, beta, mem1, mem2,
                                            dst, acc, kbuf, bar, out);
  (void)in_sizes; (void)n_in; (void)out_size;
}
